// Generator_5282809774927
// MI455X (gfx1250) — matrix operand order repaired and checked by reading, not compiled, not run on hardware
//
#include <hip/hip_runtime.h>

// ---------------------------------------------------------------------------
// SNN 3-layer LIF forward, phase-decomposed per layer (time scan per layer).
// GEMMs via V_WMMA_F32_16X16X32_F16 (gfx1250 / CDNA5, wave32).
//
// Round-3 additions:
//  - double-buffered activation tile, staged for timestep t+1 with
//    GLOBAL_LOAD_ASYNC_TO_LDS_B128 (ASYNCcnt) while computing timestep t:
//    hides all staging latency (phase 3 runs ~1 WG/WGP, nothing else hides it)
//  - A_STRIDE = Kpad+8 so LDS rows are 16B-aligned for async b128 staging
// ---------------------------------------------------------------------------

#define USE_ASYNC_STAGE 1

typedef __attribute__((ext_vector_type(16))) _Float16 v16h;
typedef __attribute__((ext_vector_type(2)))  _Float16 h2;
typedef __attribute__((ext_vector_type(8)))  float    v8f;

#define BETA  0.95f
#define THR   1.0f
#define SCALE 2.0f

// ---------------------------------------------------------------------------
// Pack W[N][K] (f32 row-major) -> f16 B-fragment blocks, dword-interleaved:
// block (kt,nt) = 256 dwords; dword (p,lane) at p*32+lane holds the f16 K-pair
// (2p, 2p+1) for lane's (half,col).  B order used here (contiguous k, not the ISA table): lanes 0-15
// hold K=0..15, lanes 16-31 hold K=16..31 of the k-tile; column N = nt*16 +
// (lane&15); VGPR p holds K pair (2p,2p+1) within the half.
// ---------------------------------------------------------------------------
__global__ void pack_weights_kernel(const float* __restrict__ W,
                                    _Float16* __restrict__ P,
                                    int K, int N, int ktiles, int ntiles) {
  int gid = blockIdx.x * blockDim.x + threadIdx.x;
  int total = ktiles * ntiles * 512;
  if (gid >= total) return;
  int blk    = gid >> 9;
  int within = gid & 511;
  int p    = within >> 6;
  int r    = within & 63;
  int lane = r >> 1;
  int pair = r & 1;
  int kt = blk / ntiles;
  int nt = blk - kt * ntiles;
  int half = lane >> 4, col = lane & 15;
  int k = kt * 32 + half * 16 + 2 * p + pair;
  int n = nt * 16 + col;
  float val = (k < K && n < N) ? W[(size_t)n * K + k] : 0.0f;
  P[gid] = (_Float16)val;
}

// ---------------------------------------------------------------------------

__device__ __forceinline__ v8f wmma_f16(v16h a, v16h b, v8f c) {
  return __builtin_amdgcn_wmma_f32_16x16x32_f16(false, a, false, b, (short)0, c,
                                                false, false);
}

// A fragment (16-bit A 16x32 layout): lanes 0-15 rows M with k
// 0..15; lanes 16-31 k 16..31 (contiguous, matches packed B).
__device__ __forceinline__ v16h load_a(const _Float16* ap, int half) {
  v16h a;
#pragma unroll
  for (int v = 0; v < 4; ++v) {
    h2 p0 = *(const h2*)(ap + half * 16 + 2 * v);
    a[2 * v]     = p0.x;
    a[2 * v + 1] = p0.y;
    h2 p1 = *(const h2*)(ap + half * 16 + 8 + 2 * v);
    a[2 * v + 8] = p1.x;
    a[2 * v + 9] = p1.y;
  }
  return a;
}

// B fragment from LDS (dword-interleaved): 8 conflict-free ds_load_b32.
__device__ __forceinline__ v16h load_b(const unsigned* base /* +lane */) {
  v16h b;
#pragma unroll
  for (int p = 0; p < 8; ++p) {
    h2 pr = *(const h2*)(base + p * 32);
    b[2 * p]     = pr.x;
    b[2 * p + 1] = pr.y;
  }
  return b;
}

#if USE_ASYNC_STAGE
// CDNA5 async copy global -> LDS, 16B per lane, tracked by ASYNCcnt.
__device__ __forceinline__ void async_g2l_b128(unsigned lds_off, const void* g) {
  asm volatile("global_load_async_to_lds_b128 %0, %1, off"
               :: "v"(lds_off), "v"((unsigned long long)(size_t)g)
               : "memory");
}
__device__ __forceinline__ void wait_async0() {
  asm volatile("s_wait_asynccnt 0x0" ::: "memory");
}
#endif

// LIF epilogue for one 16x16 tile held in a v8f accumulator.
template <bool OUT_F32>
__device__ __forceinline__ void lif_tile(
    const v8f& acc, int rtile, int ntl, int nt_begin, int mrow, int half,
    int M_STRIDE, float* __restrict__ m_tile, const float* __restrict__ bias,
    int N, void* __restrict__ out_v, int ld_out, int T, int t, int bbase) {
  const int ncol = (nt_begin + ntl) * 16 + mrow;
  const float bval = (ncol < N) ? bias[ncol] : 0.0f;
  const int clocal = ntl * 16 + mrow;
#pragma unroll
  for (int v = 0; v < 8; ++v) {
    const int rlocal = rtile + v + 8 * half;
    float* mp = &m_tile[rlocal * M_STRIDE + clocal];
    float mo = *mp;
    float c  = fmaxf(0.0f, SCALE * (acc[v] + bval));
    float mn = BETA * mo + c - ((mo > THR) ? THR : 0.0f);
    *mp = mn;
    float spk = (mn > THR) ? 1.0f : 0.0f;
    if (ncol < ld_out) {
      size_t oidx = ((size_t)(bbase + rlocal) * T + t) * ld_out + ncol;
      if (OUT_F32) ((float*)out_v)[oidx] = spk;
      else         ((_Float16*)out_v)[oidx] = (_Float16)spk;
    }
  }
}

// ---------------------------------------------------------------------------
// One SNN layer phase.  Grid: x = B/64 batch tiles, y = N-tile ranges of NTW.
// 256 threads = 8 waves; wave = (rg, ng): rows rg*32..+31, N-tile pairs
// ntp = ng, ng+4, ...  Membrane persists in LDS across the T scan.
// Activation tile double-buffered; t+1 staged async while t computes.
// ---------------------------------------------------------------------------
template <bool IN_F32, bool OUT_F32, int KTILES, int NTW>
__global__ __launch_bounds__(256) void snn_layer_kernel(
    const void* __restrict__ in_v,
    const _Float16* __restrict__ Wp, const float* __restrict__ bias,
    int K_in, int ld_in, int N, int ntiles,
    void* __restrict__ out_v, int ld_out, int T) {
  constexpr int BTILE    = 64;
  constexpr int Kpad     = KTILES * 32;
  constexpr int A_STRIDE = Kpad + 8;        // rows 16B-aligned (async b128 dst)
  constexpr int M_STRIDE = NTW * 16 + 4;    // 8*stride%64==32 -> halves disjoint
  constexpr int WBLK     = KTILES * NTW;
  constexpr int NPAIR    = NTW / 2;
  constexpr size_t ABYTES = (size_t)BTILE * A_STRIDE * 2;  // multiple of 16

  extern __shared__ char smem[];
  unsigned* w_lds  = (unsigned*)smem;
  _Float16* a_buf0 = (_Float16*)(smem + (size_t)WBLK * 1024);
  _Float16* a_buf1 = (_Float16*)((char*)a_buf0 + ABYTES);
  float*    m_tile = (float*)((char*)a_buf1 + ABYTES);

  const int tid  = threadIdx.x;
  const int lane = tid & 31, wave = tid >> 5;
  const int rg = wave & 1, ng = wave >> 1;
  const int rbase = rg * 32;
  const int bbase = blockIdx.x * BTILE;
  const int nt_begin = blockIdx.y * NTW;
  const int half = lane >> 4, mrow = lane & 15;

  // ---- stage T-invariant packed weights into LDS (zero beyond ntiles) ----
  {
    const unsigned* Wd = (const unsigned*)Wp;
    for (int i = tid; i < WBLK * 256; i += 256) {
      int kt  = i / (NTW * 256);
      int rem = i - kt * (NTW * 256);
      int ntl = rem >> 8;
      int d   = rem & 255;
      int ntg = nt_begin + ntl;
      w_lds[i] = (ntg < ntiles) ? Wd[(size_t)(kt * ntiles + ntg) * 256 + d] : 0u;
    }
  }
  for (int i = tid; i < BTILE * M_STRIDE; i += 256) m_tile[i] = 0.0f;

  // ---- stage(t, buf): issue activation staging for one timestep ----
  auto stage = [&](int t, _Float16* buf) {
    if (IN_F32) {
      // f32 -> f16 conversion path (layer 1): synchronous
      const float* in = (const float*)in_v;
      for (int idx = tid; idx < BTILE * Kpad; idx += 256) {
        int r = idx / Kpad, k = idx - r * Kpad;
        _Float16 v = (_Float16)0.0f;
        if (k < K_in) v = (_Float16)in[((size_t)(bbase + r) * T + t) * ld_in + k];
        buf[r * A_STRIDE + k] = v;
      }
    } else {
#if USE_ASYNC_STAGE
      // async DMA: rows are Kpad f16 = multiple of 16B in global and LDS
      constexpr int CPR = Kpad / 8;        // 16B chunks per row
      const char* inb = (const char*)in_v;
      unsigned lbase = (unsigned)(size_t)buf;   // low 32 bits = LDS offset
      for (int c = tid; c < BTILE * CPR; c += 256) {
        int r = c / CPR, j = c - r * CPR;
        const char* g = inb + (((size_t)(bbase + r) * T + t) * Kpad) * 2 + j * 16;
        async_g2l_b128(lbase + (unsigned)(r * (A_STRIDE * 2) + j * 16), g);
      }
#else
      const unsigned* in = (const unsigned*)in_v;
      unsigned* ad = (unsigned*)buf;
      constexpr int Kdw = Kpad / 2;
      for (int idx = tid; idx < BTILE * Kdw; idx += 256) {
        int r = idx / Kdw, kd = idx - r * Kdw;
        ad[r * (A_STRIDE / 2) + kd] = in[((size_t)(bbase + r) * T + t) * Kdw + kd];
      }
#endif
    }
  };
  auto stage_wait = [&]() {
#if USE_ASYNC_STAGE
    if (!IN_F32) wait_async0();
#endif
  };

  // ---- pipelined T scan ----
  stage(0, a_buf0);
  stage_wait();
  __syncthreads();

  for (int t = 0; t < T; ++t) {
    _Float16* cur = (t & 1) ? a_buf1 : a_buf0;
    _Float16* nxt = (t & 1) ? a_buf0 : a_buf1;
    if (t + 1 < T) stage(t + 1, nxt);   // overlaps with compute below

    for (int ntp = ng; ntp < NPAIR; ntp += 4) {
      const int ntl0 = 2 * ntp, ntl1 = ntl0 + 1;
      v8f acc00 = {}, acc01 = {}, acc10 = {}, acc11 = {};
      const _Float16* ar0 = cur + (rbase + mrow) * A_STRIDE;
      const _Float16* ar1 = ar0 + 16 * A_STRIDE;
      const unsigned* wb0 = w_lds + ntl0 * 256 + lane;
      const unsigned* wb1 = wb0 + 256;
#pragma unroll
      for (int kt = 0; kt < KTILES; ++kt) {
        v16h a0 = load_a(ar0 + kt * 32, half);
        v16h a1 = load_a(ar1 + kt * 32, half);
        v16h b0 = load_b(wb0 + kt * (NTW * 256));
        v16h b1 = load_b(wb1 + kt * (NTW * 256));
        acc00 = wmma_f16(a0, b0, acc00);
        acc01 = wmma_f16(a0, b1, acc01);
        acc10 = wmma_f16(a1, b0, acc10);
        acc11 = wmma_f16(a1, b1, acc11);
      }
      lif_tile<OUT_F32>(acc00, rbase,      ntl0, nt_begin, mrow, half, M_STRIDE,
                        m_tile, bias, N, out_v, ld_out, T, t, bbase);
      lif_tile<OUT_F32>(acc01, rbase,      ntl1, nt_begin, mrow, half, M_STRIDE,
                        m_tile, bias, N, out_v, ld_out, T, t, bbase);
      lif_tile<OUT_F32>(acc10, rbase + 16, ntl0, nt_begin, mrow, half, M_STRIDE,
                        m_tile, bias, N, out_v, ld_out, T, t, bbase);
      lif_tile<OUT_F32>(acc11, rbase + 16, ntl1, nt_begin, mrow, half, M_STRIDE,
                        m_tile, bias, N, out_v, ld_out, T, t, bbase);
    }

    stage_wait();      // this wave's async writes to nxt are complete
    __syncthreads();   // all waves done staging nxt + done reading cur
  }
}

// ---------------------------------------------------------------------------

static size_t phase_smem(int ktiles, int ntw) {
  size_t w = (size_t)ktiles * ntw * 1024;
  size_t a = (size_t)64 * (ktiles * 32 + 8) * 2;
  size_t m = (size_t)64 * (ntw * 16 + 4) * 4;
  return w + 2 * a + m;
}

extern "C" void kernel_launch(void* const* d_in, const int* in_sizes, int n_in,
                              void* d_out, int out_size, void* d_ws, size_t ws_size,
                              hipStream_t stream) {
  (void)in_sizes; (void)n_in; (void)out_size; (void)ws_size;
  const float* x  = (const float*)d_in[0];
  const float* W1 = (const float*)d_in[1];
  const float* b1 = (const float*)d_in[2];
  const float* W2 = (const float*)d_in[3];
  const float* b2 = (const float*)d_in[4];
  const float* W3 = (const float*)d_in[5];
  const float* b3 = (const float*)d_in[6];

  const int B = 8192, T = 25, D0 = 100, H1 = 100, H2 = 400, O = 784;
  const int kt1 = 4,  nt1 = 7;   // L1: K 100->128, N 100 (stored 128)
  const int kt2 = 4,  nt2 = 25;  // L2: K 128,      N 400 (stored 416)
  const int kt3 = 13, nt3 = 49;  // L3: K 416,      N 784 (== d_out)

  // workspace: packed W1|W2|W3 (f16), s1 [B,T,128] f16, s2 [B,T,416] f16
  _Float16* P1 = (_Float16*)d_ws;
  _Float16* P2 = P1 + (size_t)kt1 * nt1 * 512;
  _Float16* P3 = P2 + (size_t)kt2 * nt2 * 512;
  _Float16* S1 = P3 + (size_t)kt3 * nt3 * 512;
  _Float16* S2 = S1 + (size_t)B * T * 128;

  {
    int tot1 = kt1 * nt1 * 512;
    pack_weights_kernel<<<(tot1 + 255) / 256, 256, 0, stream>>>(W1, P1, D0, H1, kt1, nt1);
    int tot2 = kt2 * nt2 * 512;
    pack_weights_kernel<<<(tot2 + 255) / 256, 256, 0, stream>>>(W2, P2, H1, H2, kt2, nt2);
    int tot3 = kt3 * nt3 * 512;
    pack_weights_kernel<<<(tot3 + 255) / 256, 256, 0, stream>>>(W3, P3, H2, O, kt3, nt3);
  }

  // --- phase 1: x (f32,[B,T,100]) -> s1 spikes (f16, ld 128) ---
  {
    dim3 grid(B / 64, 1);
    snn_layer_kernel<true, false, 4, 8>
        <<<grid, 256, phase_smem(4, 8), stream>>>(
            x, P1, b1, D0, D0, H1, nt1, S1, 128, T);
  }
  // --- phase 2: s1 -> s2 spikes (f16, ld 416) ---
  {
    dim3 grid(B / 64, 3);
    snn_layer_kernel<false, false, 4, 10>
        <<<grid, 256, phase_smem(4, 10), stream>>>(
            S1, P2, b2, 128, 128, H2, nt2, S2, 416, T);
  }
  // --- phase 3: s2 -> output spikes (f32, [B,T,784]) ---
  {
    dim3 grid(B / 64, 5);
    snn_layer_kernel<false, true, 13, 10>
        <<<grid, 256, phase_smem(13, 10), stream>>>(
            S2, P3, b3, 416, 416, O, nt3, d_out, O, T);
  }
}
